// IndexerOp_87411174408496
// MI455X (gfx1250) — 8-bit matrix operand order repaired and checked by reading, compiled, not run on hardware
//
#include <hip/hip_runtime.h>
#include <hip/hip_bf16.h>
#include <stdint.h>

#define T_TOK  4096
#define NH     16
#define TOPK   2048
#define FP8MAX 448.0f

typedef int    v16i __attribute__((ext_vector_type(16)));
typedef int    v4i  __attribute__((ext_vector_type(4)));
typedef float  v8f  __attribute__((ext_vector_type(8)));
typedef float  v4f  __attribute__((ext_vector_type(4)));

#if __has_builtin(__builtin_amdgcn_global_load_async_to_lds_b32)
#define USE_ASYNC_LDS 1
#else
#define USE_ASYNC_LDS 0
#endif

__device__ __forceinline__ void wait_asynccnt0() {
#if __has_builtin(__builtin_amdgcn_s_wait_asynccnt)
  __builtin_amdgcn_s_wait_asynccnt(0);
#else
  asm volatile("s_wait_asynccnt 0x0" ::: "memory");
#endif
}

__device__ __forceinline__ unsigned char f32_to_e4m3_sw(float f) {
  unsigned int u = __float_as_uint(f);
  unsigned int s = (u >> 24) & 0x80u;
  float a = fabsf(f);
  if (!(a > 0.f)) return (unsigned char)s;
  if (a >= FP8MAX) return (unsigned char)(s | 0x7E);
  int e = (int)floorf(log2f(a));
  if (e < -6) e = -6;
  float q = ldexpf(rintf(ldexpf(a, 3 - e)), e - 3);
  e = (int)floorf(log2f(q)); if (e < -6) e = -6;
  int mant = (int)rintf(ldexpf(q, 3 - e)) - 8;
  if (mant < 0) { mant = (int)rintf(ldexpf(q, 9)); return (unsigned char)(s | mant); }
  if (mant > 7) { mant = 0; e += 1; }
  if (e > 8) return (unsigned char)(s | 0x7E);
  return (unsigned char)(s | ((unsigned)(e + 7) << 3) | (unsigned)mant);
}

template <bool IS_Q>
__global__ __launch_bounds__(128) void prep_kernel(
    const __hip_bfloat16* __restrict__ x,
    const __hip_bfloat16* __restrict__ w,
    const float* __restrict__ cs,
    const int* __restrict__ positions,
    uint8_t* __restrict__ out8,
    float* __restrict__ outScale)
{
  __shared__ float buf[128];
  const int row = blockIdx.x;
  const int t   = IS_Q ? (row >> 4) : row;
  const int d   = threadIdx.x;
  const __hip_bfloat16* xr = x + ((size_t)row << 7);
  const float* csr = cs + (size_t)positions[t] * 64;

  float xv = __bfloat162float(xr[d]);
  float y;
  if (d < 32) {
    float x2 = __bfloat162float(xr[d + 32]);
    y = xv * csr[d] - x2 * csr[32 + d];
  } else if (d < 64) {
    int j = d - 32;
    float x1 = __bfloat162float(xr[j]);
    y = xv * csr[j] + x1 * csr[32 + j];
  } else {
    y = xv;
  }

  buf[d] = y; __syncthreads();
  #pragma unroll
  for (int s = 1; s < 128; s <<= 1) {
    float a = buf[d], b = buf[d ^ s];
    __syncthreads();
    buf[d] = (d & s) ? (b - a) : (a + b);
    __syncthreads();
  }
  y = buf[d] * 0.08838834764831845f;
  y = __bfloat162float(__float2bfloat16(y));

  buf[d] = fabsf(y); __syncthreads();
  #pragma unroll
  for (int s = 64; s >= 1; s >>= 1) {
    if (d < s) buf[d] = fmaxf(buf[d], buf[d + s]);
    __syncthreads();
  }
  float amax  = fmaxf(buf[0], 1e-4f);
  float scale = exp2f(ceilf(log2f(amax / FP8MAX)));
  float qv    = fminf(fmaxf(y / scale, -FP8MAX), FP8MAX);

#if __has_builtin(__builtin_amdgcn_cvt_pk_fp8_f32)
  int pk = __builtin_amdgcn_cvt_pk_fp8_f32(qv, qv, 0, false);
  unsigned char byte = (unsigned char)(pk & 0xFF);
#else
  unsigned char byte = f32_to_e4m3_sw(qv);
#endif
  out8[((size_t)row << 7) + d] = byte;

  if (d == 0) {
    if (IS_Q) {
      int h = row & 15;
      outScale[row] = __bfloat162float(w[(size_t)t * NH + h]) * scale;
    } else {
      outScale[row] = scale;
    }
  }
}

__global__ __launch_bounds__(256) void gemm_kernel(
    const uint8_t* __restrict__ qf8,
    const uint8_t* __restrict__ kf8,
    const float*  __restrict__ coeff,
    const float*  __restrict__ sk,
    float* __restrict__ logits)
{
  __shared__ int   ldsA[16 * 512];
  __shared__ float ldsC[16 * 16];
  const int tid = threadIdx.x;
  const int t0  = blockIdx.x << 4;

  for (int dd = tid; dd < 8192; dd += 256) {
    int h    = dd >> 9;
    int rem  = dd & 511;
    int lane = rem >> 4;
    int v    = rem & 15;
    int M  = lane & 15, hi = lane >> 4, v7 = v & 7;
    int K  = ((v >> 2) << 5) + ((v & 3) << 2) + (hi << 4);
    size_t goff = ((((size_t)(t0 + M)) * NH + h) << 7) + (size_t)K;
#if USE_ASYNC_LDS
    int* gsrc = (int*)(void*)(qf8 + goff);
    __builtin_amdgcn_global_load_async_to_lds_b32(
        (__attribute__((address_space(1))) int*)gsrc,
        (__attribute__((address_space(3))) int*)&ldsA[dd],
        0, 0);
#else
    ldsA[dd] = *(const int*)(qf8 + goff);
#endif
  }
  { int h = tid >> 4, m = tid & 15;
    ldsC[(h << 4) + m] = coeff[((size_t)(t0 + m)) * NH + h]; }
#if USE_ASYNC_LDS
  wait_asynccnt0();
#endif
  __syncthreads();

  const int wave = tid >> 5, lane = tid & 31;
  const int n = lane & 15, hi = lane >> 4;
  const int nTiles = (t0 >> 4) + 1;

  for (int st = wave; st < nTiles; st += 8) {
    const int s0 = st << 4;
    const uint8_t* krow = kf8 + (((size_t)(s0 + n)) << 7) + (hi << 4);
    if (st + 8 < nTiles)
      __builtin_prefetch(kf8 + (((size_t)(s0 + 128 + n)) << 7), 0, 0);

    v16i b;
    #pragma unroll
    for (int c = 0; c < 4; ++c)
      ((v4i*)&b)[c] = *(const v4i*)(krow + c * 32);

    float part[8];
    #pragma unroll
    for (int i = 0; i < 8; ++i) part[i] = 0.f;

    #pragma unroll 4
    for (int h = 0; h < 16; ++h) {
      v16i a;
      const v4i* ap = (const v4i*)&ldsA[(h << 9) + (lane << 4)];
      #pragma unroll
      for (int c = 0; c < 4; ++c) ((v4i*)&a)[c] = ap[c];

      v8f acc = {};
      acc = __builtin_amdgcn_wmma_f32_16x16x128_fp8_fp8(a, b, (short)0, acc,
                                                        false, false);
      const v4f* cp = (const v4f*)&ldsC[(h << 4) + (hi << 3)];
      v4f c0 = cp[0], c1 = cp[1];
      #pragma unroll
      for (int i = 0; i < 4; ++i) {
        part[i]     += c0[i] * fmaxf(acc[i], 0.f);
        part[i + 4] += c1[i] * fmaxf(acc[i + 4], 0.f);
      }
    }
    const float skv = sk[s0 + n];
    #pragma unroll
    for (int i = 0; i < 8; ++i) {
      int   m   = t0 + i + (hi << 3);
      int   nn  = s0 + n;
      float val = part[i] * skv;
      if (nn > m) val = -1e30f;
      logits[(size_t)m * T_TOK + nn] = val;
    }
  }
}

__global__ __launch_bounds__(256) void topk_kernel(
    const float* __restrict__ logits,
    float* __restrict__ outScores, int* __restrict__ outIdx)
{
  __shared__ unsigned long long keys[T_TOK];
  const int t = blockIdx.x, tid = threadIdx.x;
  const float* row = logits + (size_t)t * T_TOK;

  for (int s = tid; s < T_TOK; s += 256) {
    float v = (s <= t) ? row[s] : -1e30f;
    unsigned int u = __float_as_uint(v);
    u = (u & 0x80000000u) ? ~u : (u | 0x80000000u);
    keys[s] = ((unsigned long long)u << 32) |
              (unsigned long long)(0xFFFFFFFFu - (unsigned int)s);
  }
  __syncthreads();

  for (int k = 2; k <= T_TOK; k <<= 1) {
    for (int j = k >> 1; j > 0; j >>= 1) {
      for (int idx = tid; idx < T_TOK; idx += 256) {
        int ixj = idx ^ j;
        if (ixj > idx) {
          unsigned long long a = keys[idx], b = keys[ixj];
          if ((a < b) == ((idx & k) == 0)) {
            keys[idx] = b; keys[ixj] = a;
          }
        }
      }
      __syncthreads();
    }
  }

  for (int i = tid; i < TOPK; i += 256) {
    unsigned long long kk = keys[i];
    unsigned int uk = (unsigned int)(kk >> 32);
    float sc = (uk & 0x80000000u) ? __uint_as_float(uk ^ 0x80000000u)
                                  : __uint_as_float(~uk);
    outScores[(size_t)t * TOPK + i] = sc;
    outIdx  [(size_t)t * TOPK + i] =
        (int)(0xFFFFFFFFu - (unsigned int)(kk & 0xFFFFFFFFull));
  }
}

extern "C" void kernel_launch(void* const* d_in, const int* in_sizes, int n_in,
                              void* d_out, int out_size, void* d_ws, size_t ws_size,
                              hipStream_t stream) {
  (void)in_sizes; (void)n_in; (void)out_size; (void)ws_size;
  const __hip_bfloat16* q  = (const __hip_bfloat16*)d_in[0];
  const __hip_bfloat16* k  = (const __hip_bfloat16*)d_in[1];
  const __hip_bfloat16* w  = (const __hip_bfloat16*)d_in[2];
  const float*          cs = (const float*)d_in[3];
  const int*            ps = (const int*)d_in[4];

  uint8_t* ws  = (uint8_t*)d_ws;
  size_t   off = 0;
  auto alloc = [&](size_t bytes) -> void* {
    void* p = ws + off;
    off = (off + bytes + 255) & ~(size_t)255;
    return p;
  };
  uint8_t* qf8    = (uint8_t*)alloc((size_t)T_TOK * NH * 128);
  uint8_t* kf8    = (uint8_t*)alloc((size_t)T_TOK * 128);
  float*   coeff  = (float*)  alloc((size_t)T_TOK * NH * sizeof(float));
  float*   sk     = (float*)  alloc((size_t)T_TOK * sizeof(float));
  float*   logits = (float*)  alloc((size_t)T_TOK * T_TOK * sizeof(float));

  prep_kernel<true ><<<T_TOK * NH, 128, 0, stream>>>(q, w, cs, ps, qf8, coeff);
  prep_kernel<false><<<T_TOK,      128, 0, stream>>>(k, nullptr, cs, ps, kf8, sk);
  gemm_kernel<<<T_TOK / 16, 256, 0, stream>>>(qf8, kf8, coeff, sk, logits);

  float* outScores = (float*)d_out;
  int*   outIdx    = (int*)((float*)d_out + (size_t)T_TOK * TOPK);
  topk_kernel<<<T_TOK, 256, 0, stream>>>(logits, outScores, outIdx);
}
